// my_model_31688268710358
// MI455X (gfx1250) — matrix operand order repaired and checked by reading, not compiled, not run on hardware
//
#include <hip/hip_runtime.h>

// ---------------------------------------------------------------------------
// CDNA5 (gfx1250) wave32 WMMA types & helpers
// ---------------------------------------------------------------------------
typedef __attribute__((ext_vector_type(16))) __bf16          v16bf;
typedef __attribute__((ext_vector_type(8)))  float           v8f;
typedef __attribute__((ext_vector_type(8)))  unsigned short  v8us;

union BF16x16 { v16bf v; v8us u[2]; unsigned short s[16]; };

__device__ __forceinline__ unsigned short f2bf_us(float f) {
  unsigned u = __builtin_bit_cast(unsigned, f);
  unsigned r = u + 0x7FFFu + ((u >> 16) & 1u);   // RTNE
  return (unsigned short)(r >> 16);
}
__device__ __forceinline__ float bf2f(unsigned short s) {
  unsigned u = ((unsigned)s) << 16;
  return __builtin_bit_cast(float, u);
}
__device__ __forceinline__ float sigm(float x) { return 1.f / (1.f + __expf(-x)); }
__device__ __forceinline__ float tanh_fast(float x) {
  x = fminf(fmaxf(x, -15.f), 15.f);
  float e = __expf(2.f * x);
  return (e - 1.f) / (e + 1.f);
}

__device__ __forceinline__ v8f wmma_bf16(v16bf a, v16bf b, v8f c) {
  return __builtin_amdgcn_wmma_f32_16x16x32_bf16(false, a, false, b, (short)0, c,
                                                 false, false);
}

// ------------------- CDNA5 async global<->LDS DMA (ASYNCcnt) ----------------
typedef __attribute__((address_space(3))) const void lds_cv;
__device__ __forceinline__ unsigned lds_byte_off(const void* p) {
  return (unsigned)(unsigned long long)(lds_cv*)p;
}
__device__ __forceinline__ void async_ld_b128(unsigned ldsByte, const void* g) {
  asm volatile("global_load_async_to_lds_b128 %0, %1, off"
               :: "v"(ldsByte), "v"((unsigned long long)g) : "memory");
}
__device__ __forceinline__ void async_st_b128(void* g, unsigned ldsByte) {
  asm volatile("global_store_async_from_lds_b128 %0, %1, off"
               :: "v"((unsigned long long)g), "v"(ldsByte) : "memory");
}
#define WAIT_ASYNC0()  asm volatile("s_wait_asynccnt 0x0"  ::: "memory")
#define WAIT_ASYNC16() asm volatile("s_wait_asynccnt 0x10" ::: "memory")
#define WAIT_DS0()     asm volatile("s_wait_dscnt 0x0"     ::: "memory")

// ------------------- WMMA operand loaders (ISA 7.12.2 layouts) ---------------
// A: 16x32 bf16 from [16][64] tile, stride 64. kh = K-half.
__device__ __forceinline__ v16bf load_A16(const unsigned short* h, int lane, int kh) {
  int row = lane & 15;
  int kb  = ((lane < 16) ? 0 : 8) + kh * 32;
  const unsigned short* p = h + row * 64 + kb;
  BF16x16 a;
  a.u[0] = *(const v8us*)p;          // K = kb .. kb+7
  a.u[1] = *(const v8us*)(p + 16);   // K = kb+16 .. kb+23
  return a.v;
}
// B: 32x16 bf16 with B(k,n) = W[n][k], W row-major stride 64.
__device__ __forceinline__ v16bf load_B16(const unsigned short* w, int nt, int kh, int lane) {
  int n  = nt * 16 + (lane & 15);
  int kb = ((lane < 16) ? 0 : 8) + kh * 32;
  const unsigned short* p = w + n * 64 + kb;
  BF16x16 b;
  b.u[0] = *(const v8us*)p;
  b.u[1] = *(const v8us*)(p + 16);
  return b.v;
}
// B for K-padded input weight (stored [192][16], K>=6 zero).
__device__ __forceinline__ v16bf load_B_wp(const unsigned short* w, int nt, int lane) {
  BF16x16 b;
  if (lane < 16) {
    const unsigned short* p = w + (nt * 16 + lane) * 16;
    b.u[0] = *(const v8us*)p;
    b.u[1] = *(const v8us*)(p + 8);
  } else {
#pragma unroll
    for (int i = 0; i < 16; ++i) b.s[i] = 0;
  }
  return b.v;
}
// A for the per-step input slice (K=6 valid), LDS [16][6][64].
__device__ __forceinline__ v16bf load_A_x(const unsigned short* xs, int t, int lane) {
  BF16x16 a;
#pragma unroll
  for (int i = 0; i < 16; ++i) a.s[i] = 0;
  if (lane < 16) {
    const unsigned short* p = xs + lane * 384 + t;
#pragma unroll
    for (int f = 0; f < 6; ++f) a.s[f] = p[f * 64];
  }
  return a.v;
}

// ---------------------------------------------------------------------------
// K1: normalized time-series rows (feature 4) -> bf16
// ---------------------------------------------------------------------------
__global__ void prep_xn(const float* __restrict__ X, unsigned short* __restrict__ xnb, int N) {
  int row = blockIdx.x * blockDim.x + threadIdx.x;
  if (row >= N) return;
  const float* ts = X + (size_t)row * 384 + 256;  // X[row][4][t]
  float m = 0.f;
  for (int t = 0; t < 64; ++t) m += ts[t];
  m *= (1.f / 64.f);
  float ss = 0.f;
  for (int t = 0; t < 64; ++t) { float d = ts[t] - m; ss += d * d; }
  float inv = 1.f / sqrtf(ss + 1e-12f);
  for (int t = 0; t < 64; ++t) xnb[row * 64 + t] = f2bf_us((ts[t] - m) * inv);
}

// ---------------------------------------------------------------------------
// K1b: one-time f32 -> bf16 weight image [Whh0 | Wih1 | Whh1 | Wih0 K-pad16]
// (39936 bf16 = 78 KB; every GRU workgroup then DMA's this straight to LDS)
// ---------------------------------------------------------------------------
__global__ void prep_weights(const float* __restrict__ Whh0, const float* __restrict__ Wih1,
                             const float* __restrict__ Whh1, const float* __restrict__ Wih0,
                             unsigned short* __restrict__ wbf) {
  int i = blockIdx.x * blockDim.x + threadIdx.x;
  if (i < 12288)       wbf[i] = f2bf_us(Whh0[i]);
  else if (i < 24576)  wbf[i] = f2bf_us(Wih1[i - 12288]);
  else if (i < 36864)  wbf[i] = f2bf_us(Whh1[i - 24576]);
  else if (i < 39936) {
    int j = i - 36864, n = j >> 4, k = j & 15;
    wbf[i] = (k < 6) ? f2bf_us(Wih0[n * 6 + k]) : (unsigned short)0;
  }
}

// ---------------------------------------------------------------------------
// K2: fused corr = xn @ xn^T (WMMA) + per-row top-10.
// One wave per 16 rows; column blocks of 64 double-buffered in LDS via
// global_load_async_to_lds_b128 so DMA overlaps WMMA + top-k scan.
// ---------------------------------------------------------------------------
__global__ __launch_bounds__(32) void corr_topk(const unsigned short* __restrict__ xnb,
                                                int* __restrict__ adj, int nblk) {
  __shared__ __align__(16) unsigned short bbuf[2][64 * 64];  // 2 x 8KB column blocks
  __shared__ __align__(16) float cT[16][64];
  const int lane = threadIdx.x;
  const int rowBase = blockIdx.x * 16;
  const v8f z8 = {0.f, 0.f, 0.f, 0.f, 0.f, 0.f, 0.f, 0.f};

  v16bf A0 = load_A16(xnb + (size_t)rowBase * 64, lane, 0);
  v16bf A1 = load_A16(xnb + (size_t)rowBase * 64, lane, 1);

  float tv[10]; int ti[10];
#pragma unroll
  for (int q = 0; q < 10; ++q) { tv[q] = -3.4e38f; ti[q] = 0; }

  const int hi = lane >> 4, col = lane & 15;

  { // prime block 0 (16 x 512B chunks)
    unsigned lb = lds_byte_off(bbuf[0]);
    const char* src = (const char*)xnb;
#pragma unroll
    for (int i = 0; i < 16; ++i)
      async_ld_b128(lb + i * 512 + lane * 16, src + i * 512 + lane * 16);
  }

  for (int blk = 0; blk < nblk; ++blk) {
    int nxt = blk + 1;
    if (nxt < nblk) {       // stream next block while we compute this one
      unsigned lb = lds_byte_off(bbuf[nxt & 1]);
      const char* src = (const char*)xnb + (size_t)nxt * 8192;
#pragma unroll
      for (int i = 0; i < 16; ++i)
        async_ld_b128(lb + i * 512 + lane * 16, src + i * 512 + lane * 16);
      WAIT_ASYNC16();       // in-order: <=16 outstanding => current block landed
    } else {
      WAIT_ASYNC0();
    }
    const unsigned short* bb = bbuf[blk & 1];
#pragma unroll
    for (int st = 0; st < 4; ++st) {
      v8f c = wmma_bf16(A1, load_B16(bb, st, 1, lane),
               wmma_bf16(A0, load_B16(bb, st, 0, lane), z8));
#pragma unroll
      for (int r = 0; r < 8; ++r) cT[r + 8 * hi][st * 16 + col] = c[r];
    }
    __syncthreads();                       // single-wave WG: ordering fence
    if (lane < 16) {
      int colBase = blk * 64;
      for (int j = 0; j < 64; ++j) {
        float v = cT[lane][j];
        if (v > tv[9]) {                   // rare insert into sorted top-10
          tv[9] = v; ti[9] = colBase + j;
#pragma unroll
          for (int p = 9; p > 0; --p) {
            if (tv[p] > tv[p - 1]) {
              float tf = tv[p]; tv[p] = tv[p - 1]; tv[p - 1] = tf;
              int   tt = ti[p]; ti[p] = ti[p - 1]; ti[p - 1] = tt;
            }
          }
        }
      }
    }
    __syncthreads();
  }
  if (lane < 16) {
    int row = rowBase + lane;
#pragma unroll
    for (int q = 0; q < 10; ++q) adj[row * 16 + q] = ti[q];
  }
}

// ---------------------------------------------------------------------------
// K3: fused 2-layer GRU scan. 4 waves x 16 nodes per workgroup; bf16 weights
// DMA'd to LDS (async), h0/h1 LDS-resident bf16; 64-step loop is barrier-free.
// ---------------------------------------------------------------------------
__device__ __forceinline__ v8f gru_update(v8f gxr, v8f gxz, v8f gxn,
                                          v8f ghr, v8f ghz, v8f ghn,
                                          const float* bx, const float* bh,
                                          const unsigned short* h,
                                          int rbase, int col, int c) {
  const int gcol = c * 16 + col;
  float bxr = bx[gcol], bxz = bx[64 + gcol], bxn = bx[128 + gcol];
  float bhr = bh[gcol], bhz = bh[64 + gcol], bhn = bh[128 + gcol];
  v8f out;
#pragma unroll
  for (int rv = 0; rv < 8; ++rv) {
    float r = sigm(gxr[rv] + bxr + ghr[rv] + bhr);
    float z = sigm(gxz[rv] + bxz + ghz[rv] + bhz);
    float n = tanh_fast(gxn[rv] + bxn + r * (ghn[rv] + bhn));
    float hold = bf2f(h[(rbase + rv) * 64 + gcol]);
    out[rv] = (1.f - z) * n + z * hold;
  }
  return out;
}

__global__ __launch_bounds__(128, 1) void gru_fused(
    const float* __restrict__ X, const unsigned short* __restrict__ wbf,
    const float* __restrict__ bih0, const float* __restrict__ bhh0,
    const float* __restrict__ bih1, const float* __restrict__ bhh1,
    unsigned short* __restrict__ lastb) {
  __shared__ __align__(16) unsigned short swts[39936];     // Whh0|Wih1|Whh1|Wih0p
  __shared__ __align__(16) unsigned short sX[64 * 6 * 64]; // [node][f][t]
  __shared__ __align__(16) unsigned short h0b[64 * 64];
  __shared__ __align__(16) unsigned short h1b[64 * 64];
  __shared__ float sbias[768];

  const int tid = threadIdx.x;
  const int wv = tid >> 5, lane = tid & 31;

  { // async DMA the prepacked bf16 weight image: 156 x 512B chunks, 39/wave
    unsigned lb = lds_byte_off(swts);
    const char* src = (const char*)wbf;
    for (int ci = wv; ci < 156; ci += 4)
      async_ld_b128(lb + ci * 512 + lane * 16, src + ci * 512 + lane * 16);
  }
  for (int i = tid; i < 192; i += 128) {
    sbias[i] = bih0[i]; sbias[192 + i] = bhh0[i];
    sbias[384 + i] = bih1[i]; sbias[576 + i] = bhh1[i];
  }
  const float* Xblk = X + (size_t)blockIdx.x * 64 * 384;
  for (int i = tid; i < 64 * 384; i += 128) sX[i] = f2bf_us(Xblk[i]);
  for (int i = tid; i < 64 * 64; i += 128) { h0b[i] = 0; h1b[i] = 0; }
  WAIT_ASYNC0();
  __syncthreads();

  const unsigned short* sWhh0  = swts;
  const unsigned short* sWih1  = swts + 12288;
  const unsigned short* sWhh1  = swts + 24576;
  const unsigned short* sWih0p = swts + 36864;

  unsigned short* myh0 = h0b + wv * 1024;
  unsigned short* myh1 = h1b + wv * 1024;
  const unsigned short* myX = sX + wv * 16 * 384;
  const int col = lane & 15;
  const int rbase = (lane >> 4) * 8;
  const v8f z8 = {0.f, 0.f, 0.f, 0.f, 0.f, 0.f, 0.f, 0.f};

  for (int t = 0; t < 64; ++t) {
    // ------------- layer 0 -------------
    v16bf a0 = load_A16(myh0, lane, 0);
    v16bf a1 = load_A16(myh0, lane, 1);
    v16bf ax = load_A_x(myX, t, lane);
    v8f hn0[4];
#pragma unroll
    for (int c = 0; c < 4; ++c) {
      v8f ghr = wmma_bf16(a1, load_B16(sWhh0, c,     1, lane),
                 wmma_bf16(a0, load_B16(sWhh0, c,     0, lane), z8));
      v8f ghz = wmma_bf16(a1, load_B16(sWhh0, 4 + c, 1, lane),
                 wmma_bf16(a0, load_B16(sWhh0, 4 + c, 0, lane), z8));
      v8f ghn = wmma_bf16(a1, load_B16(sWhh0, 8 + c, 1, lane),
                 wmma_bf16(a0, load_B16(sWhh0, 8 + c, 0, lane), z8));
      v8f gxr = wmma_bf16(ax, load_B_wp(sWih0p, c,     lane), z8);
      v8f gxz = wmma_bf16(ax, load_B_wp(sWih0p, 4 + c, lane), z8);
      v8f gxn = wmma_bf16(ax, load_B_wp(sWih0p, 8 + c, lane), z8);
      hn0[c] = gru_update(gxr, gxz, gxn, ghr, ghz, ghn,
                          sbias, sbias + 192, myh0, rbase, col, c);
    }
#pragma unroll
    for (int c = 0; c < 4; ++c)
#pragma unroll
      for (int rv = 0; rv < 8; ++rv)
        myh0[(rbase + rv) * 64 + c * 16 + col] = f2bf_us(hn0[c][rv]);

    // ------------- layer 1 -------------
    v16bf b0 = load_A16(myh0, lane, 0);   // new h0 (same wave just wrote it)
    v16bf b1 = load_A16(myh0, lane, 1);
    v16bf c0 = load_A16(myh1, lane, 0);
    v16bf c1 = load_A16(myh1, lane, 1);
    v8f hn1[4];
#pragma unroll
    for (int c = 0; c < 4; ++c) {
      v8f gxr = wmma_bf16(b1, load_B16(sWih1, c,     1, lane),
                 wmma_bf16(b0, load_B16(sWih1, c,     0, lane), z8));
      v8f gxz = wmma_bf16(b1, load_B16(sWih1, 4 + c, 1, lane),
                 wmma_bf16(b0, load_B16(sWih1, 4 + c, 0, lane), z8));
      v8f gxn = wmma_bf16(b1, load_B16(sWih1, 8 + c, 1, lane),
                 wmma_bf16(b0, load_B16(sWih1, 8 + c, 0, lane), z8));
      v8f ghr = wmma_bf16(c1, load_B16(sWhh1, c,     1, lane),
                 wmma_bf16(c0, load_B16(sWhh1, c,     0, lane), z8));
      v8f ghz = wmma_bf16(c1, load_B16(sWhh1, 4 + c, 1, lane),
                 wmma_bf16(c0, load_B16(sWhh1, 4 + c, 0, lane), z8));
      v8f ghn = wmma_bf16(c1, load_B16(sWhh1, 8 + c, 1, lane),
                 wmma_bf16(c0, load_B16(sWhh1, 8 + c, 0, lane), z8));
      hn1[c] = gru_update(gxr, gxz, gxn, ghr, ghz, ghn,
                          sbias + 384, sbias + 576, myh1, rbase, col, c);
    }
#pragma unroll
    for (int c = 0; c < 4; ++c)
#pragma unroll
      for (int rv = 0; rv < 8; ++rv)
        myh1[(rbase + rv) * 64 + c * 16 + col] = f2bf_us(hn1[c][rv]);
  }

  // push the wave's final hidden tile (2KB) out via async store-from-LDS
  WAIT_DS0();
  char* dst = (char*)(lastb + ((size_t)blockIdx.x * 64 + wv * 16) * 64);
  unsigned lsrc = lds_byte_off(myh1);
#pragma unroll
  for (int c = 0; c < 4; ++c)
    async_st_b128(dst + c * 512 + lane * 16, lsrc + c * 512 + lane * 16);
  WAIT_ASYNC0();
}

// ---------------------------------------------------------------------------
// K4a: Xg = last @ W_gat^T + b_gat (WMMA) -> reduce to e1, e2, s = Xg.w_out
// ---------------------------------------------------------------------------
__global__ __launch_bounds__(32) void gat_feats(
    const unsigned short* __restrict__ lastb,
    const float* __restrict__ Wg, const float* __restrict__ bg,
    const float* __restrict__ we1, const float* __restrict__ be1,
    const float* __restrict__ we2, const float* __restrict__ be2,
    const float* __restrict__ wo,
    float* __restrict__ e1g, float* __restrict__ e2g, float* __restrict__ sg) {
  __shared__ __align__(16) unsigned short sWg[64 * 64];
  __shared__ __align__(16) float tile[16][64];
  __shared__ float sb[64];
  const int lane = threadIdx.x;
  for (int i = lane; i < 4096; i += 32) sWg[i] = f2bf_us(Wg[i]);
  for (int i = lane; i < 64; i += 32) sb[i] = bg[i];
  __syncthreads();

  const int rowBase = blockIdx.x * 16;
  const v8f z8 = {0.f, 0.f, 0.f, 0.f, 0.f, 0.f, 0.f, 0.f};
  v16bf A0 = load_A16(lastb + (size_t)rowBase * 64, lane, 0);
  v16bf A1 = load_A16(lastb + (size_t)rowBase * 64, lane, 1);
  const int hi = lane >> 4, col = lane & 15;
#pragma unroll
  for (int nt = 0; nt < 4; ++nt) {
    v8f c = wmma_bf16(A1, load_B16(sWg, nt, 1, lane),
             wmma_bf16(A0, load_B16(sWg, nt, 0, lane), z8));
#pragma unroll
    for (int r = 0; r < 8; ++r) tile[r + 8 * hi][nt * 16 + col] = c[r] + sb[nt * 16 + col];
  }
  __syncthreads();
  if (lane < 16) {
    float d1 = 0.f, d2 = 0.f, d3 = 0.f;
    for (int k = 0; k < 64; ++k) {
      float x = tile[lane][k];
      d1 += x * we1[k]; d2 += x * we2[k]; d3 += x * wo[k];
    }
    int row = rowBase + lane;
    e1g[row] = d1 + be1[0];
    e2g[row] = d2 + be2[0];
    sg[row]  = d3;
  }
}

// ---------------------------------------------------------------------------
// K4b: sparse GAT softmax + output: out_i = sum_j att_ij * s_j + b_out
// ---------------------------------------------------------------------------
__global__ void gat_out(const int* __restrict__ adj,
                        const float* __restrict__ e1g, const float* __restrict__ e2g,
                        const float* __restrict__ sg, const float* __restrict__ bo,
                        float* __restrict__ out, int N) {
  int i = blockIdx.x * blockDim.x + threadIdx.x;
  if (i >= N) return;
  float e1 = e1g[i];
  float ev[10], sv[10];
  float m = -3.4e38f;
#pragma unroll
  for (int q = 0; q < 10; ++q) {
    int j = adj[i * 16 + q];
    float e = e1 + e2g[j];
    e = (e > 0.f) ? e : 0.01f * e;       // leaky_relu 0.01
    ev[q] = e; sv[q] = sg[j];
    m = fmaxf(m, e);
  }
  float Z = 0.f, acc = 0.f;
#pragma unroll
  for (int q = 0; q < 10; ++q) {
    float w = __expf(ev[q] - m);
    Z += w; acc += w * sv[q];
  }
  out[i] = acc / Z + bo[0];
}

// ---------------------------------------------------------------------------
// Launch
// ---------------------------------------------------------------------------
extern "C" void kernel_launch(void* const* d_in, const int* in_sizes, int n_in,
                              void* d_out, int out_size, void* d_ws, size_t ws_size,
                              hipStream_t stream) {
  const float* X    = (const float*)d_in[0];
  const float* Wih0 = (const float*)d_in[1];
  const float* Whh0 = (const float*)d_in[2];
  const float* bih0 = (const float*)d_in[3];
  const float* bhh0 = (const float*)d_in[4];
  const float* Wih1 = (const float*)d_in[5];
  const float* Whh1 = (const float*)d_in[6];
  const float* bih1 = (const float*)d_in[7];
  const float* bhh1 = (const float*)d_in[8];
  const float* Wgat = (const float*)d_in[9];
  const float* bgat = (const float*)d_in[10];
  const float* we1  = (const float*)d_in[11];
  const float* be1  = (const float*)d_in[12];
  const float* we2  = (const float*)d_in[13];
  const float* be2  = (const float*)d_in[14];
  const float* wo   = (const float*)d_in[15];
  const float* bo   = (const float*)d_in[16];
  float* out = (float*)d_out;

  const int N = in_sizes[0] / (6 * 64);   // 8192

  char* ws = (char*)d_ws;
  unsigned short* xnb   = (unsigned short*)ws;                       // N*64 bf16
  unsigned short* lastb = (unsigned short*)(ws + (size_t)N * 128);   // N*64 bf16
  int*   adj = (int*)  (ws + (size_t)N * 256);                       // N*16 int
  float* e1g = (float*)(ws + (size_t)N * 320);                       // N f32
  float* e2g = e1g + N;
  float* sg  = e2g + N;
  unsigned short* wbf = (unsigned short*)(ws + (size_t)N * 332);     // 39936 bf16

  prep_xn     <<<(N + 255) / 256, 256, 0, stream>>>(X, xnb, N);
  prep_weights<<<(39936 + 255) / 256, 256, 0, stream>>>(Whh0, Wih1, Whh1, Wih0, wbf);
  corr_topk   <<<N / 16, 32, 0, stream>>>(xnb, adj, N / 64);
  gru_fused   <<<N / 64, 128, 0, stream>>>(X, wbf, bih0, bhh0, bih1, bhh1, lastb);
  gat_feats   <<<N / 16, 32, 0, stream>>>(lastb, Wgat, bgat, we1, be1, we2, be2, wo,
                                          e1g, e2g, sg);
  gat_out     <<<(N + 255) / 256, 256, 0, stream>>>(adj, e1g, e2g, sg, bo, out, N);
}
